// LocalRNN_18305150616306
// MI455X (gfx1250) — matrix operand order repaired and checked by reading, not compiled, not run on hardware
//
#include <hip/hip_runtime.h>

// ---------------------------------------------------------------------------
// LocalRNN (windowed GRU), B=8, L=1024, D=512, K=7 on MI455X (gfx1250).
//   1) convert x, W_ih, W_hh to f16 in workspace
//   2) gxfull = x @ W_ih^T + b_ih  (one wave per 16x16 tile, v_wmma f16->f32)
//   3) fused 7-step GRU recurrence, 64 rows/block:
//        - each wave owns a 64-wide hidden-column strip; for each 16-col tile
//          it accumulates r/z/n gate tiles in registers (B frags reused across
//          4 row tiles), then does the GRU update per-lane on the WMMA C
//          layout -- no LDS gate staging at all.
//        - h kept in LDS: f32 master (owner-lane private) + ping-pong f16
//          copy for WMMA A-fragments (read by all waves).
// ---------------------------------------------------------------------------

typedef _Float16 v8h  __attribute__((ext_vector_type(8)));
typedef _Float16 v16h __attribute__((ext_vector_type(16)));
typedef float    v8f  __attribute__((ext_vector_type(8)));

#define DHID  512
#define GD    1536          // 3*DHID
#define NROWS 8192          // B*L
#define LSEQ  1024
#define KWIN  7
#define MTILE 64            // rows per block in local_gru

__device__ __forceinline__ float fsigmoid(float v) {
  return 1.f / (1.f + __expf(-v));
}

// ---------------------------------------------------------------------------
__global__ void cvt_f32_f16(const float* __restrict__ src,
                            _Float16* __restrict__ dst, int n) {
  int i = blockIdx.x * blockDim.x + threadIdx.x;
  if (i < n) dst[i] = (_Float16)src[i];
}

// ---------------------------------------------------------------------------
// Fragment loaders per CDNA5 ISA VGPR layouts (16-bit, 16x16x32).
// A (16x32): lane m=lane&15; lanes<16 hold K {0..7,16..23}, lanes>=16 hold
//            K {8..15,24..31} -> two 16B groups.
// B (32x16): lane n=lane&15; lanes<16 hold K {0..7,16..23}, lanes>=16 hold
//            K {8..15,24..31} of W[n][k] (row-major [N][K], no transpose).
// ---------------------------------------------------------------------------
__device__ __forceinline__ v16h load_a_frag(const _Float16* base, int lda,
                                            int kk, int lane) {
  const int m    = lane & 15;
  const int aoff = (lane & 16) ? 8 : 0;
  v8h a0 = *(const v8h*)(base + m * lda + kk + aoff);
  v8h a1 = *(const v8h*)(base + m * lda + kk + 16 + aoff);
  v16h a;
#pragma unroll
  for (int i = 0; i < 8; ++i) { a[i] = a0[i]; a[i + 8] = a1[i]; }
  return a;
}

__device__ __forceinline__ v16h load_b_frag(const _Float16* __restrict__ W,
                                            int ldw, int kk, int lane) {
  const int n    = lane & 15;
  const int boff = (lane & 16) ? 8 : 0;
  v8h b0 = *(const v8h*)(W + n * ldw + kk + boff);
  v8h b1 = *(const v8h*)(W + n * ldw + kk + boff + 16);
  v16h b;
#pragma unroll
  for (int i = 0; i < 8; ++i) { b[i] = b0[i]; b[i + 8] = b1[i]; }
  return b;
}

#define WMMA_F16(a, b, c) \
  __builtin_amdgcn_wmma_f32_16x16x32_f16(false, (a), false, (b), (short)0, (c), false, false)

// ---------------------------------------------------------------------------
// gxfull[m][n] = sum_k x16[m][k] * Wih[n][k] + b_ih[n]
// grid: (GD/64, NROWS/16), block: 128 threads (4 waves, one 16x16 tile each)
// ---------------------------------------------------------------------------
__global__ void __launch_bounds__(128)
gx_gemm(const _Float16* __restrict__ xh, const _Float16* __restrict__ wih,
        const float* __restrict__ b_ih, float* __restrict__ gx) {
  const int lane = threadIdx.x & 31;
  const int wave = threadIdx.x >> 5;
  const int n0 = blockIdx.x * 64 + wave * 16;
  const int m0 = blockIdx.y * 16;

  v8f acc = {};
  const _Float16* A = xh + (size_t)m0 * DHID;
  const _Float16* W = wih + (size_t)n0 * DHID;
#pragma unroll
  for (int kk = 0; kk < DHID; kk += 32) {
    v16h a = load_a_frag(A, DHID, kk, lane);
    v16h b = load_b_frag(W, DHID, kk, lane);
    acc = WMMA_F16(a, b, acc);
  }

  const int col   = n0 + (lane & 15);
  const int rbase = m0 + ((lane & 16) ? 8 : 0);
  const float bias = b_ih[col];
#pragma unroll
  for (int r = 0; r < 8; ++r)
    gx[(size_t)(rbase + r) * GD + col] = acc[r] + bias;
}

// ---------------------------------------------------------------------------
// Fused 7-step GRU recurrence, 64 rows per 256-thread block (8 waves).
// ---------------------------------------------------------------------------
__global__ void __launch_bounds__(256)
local_gru(const _Float16* __restrict__ whh, const float* __restrict__ b_hh,
          const float* __restrict__ gx, float* __restrict__ out) {
  __shared__ float    hf[MTILE * DHID];         // 128 KB, owner-lane private
  __shared__ _Float16 hh[2][MTILE * DHID];      // 2 x 64 KB ping-pong f16 h

  const int tid   = threadIdx.x;
  const int lane  = tid & 31;
  const int wave  = tid >> 5;                   // 0..7
  const int m0    = blockIdx.x * MTILE;
  const int l0    = m0 & (LSEQ - 1);
  const int bbase = m0 & ~(LSEQ - 1);

  for (int i = tid; i < MTILE * DHID; i += 256) {
    hf[i] = 0.f;
    hh[0][i] = (_Float16)0.f;
  }
  __syncthreads();

  const int c     = lane & 15;
  const int rbase = (lane & 16) ? 8 : 0;

  int cur = 0;
  for (int k = 0; k < KWIN; ++k) {
    const int shift = k - (KWIN - 1);
    const int nxt = cur ^ 1;

    for (int dt = 0; dt < 4; ++dt) {
      const int d0 = (wave * 4 + dt) * 16;      // hidden-column tile base
      const _Float16* Wr = whh + (size_t)(d0) * DHID;
      const _Float16* Wz = whh + (size_t)(DHID + d0) * DHID;
      const _Float16* Wn = whh + (size_t)(2 * DHID + d0) * DHID;

      v8f accR[4], accZ[4], accN[4];
#pragma unroll
      for (int rt = 0; rt < 4; ++rt) {
        accR[rt] = (v8f){};
        accZ[rt] = (v8f){};
        accN[rt] = (v8f){};
      }

      // K-loop outer so the 3 B-fragments are reused across 4 row tiles.
      // Fully unrolled: lets the scheduler hoist next-iteration loads under
      // the current WMMAs (same pipelining gx_gemm already achieves).
#pragma unroll
      for (int kk = 0; kk < DHID; kk += 32) {
        v16h br = load_b_frag(Wr, DHID, kk, lane);
        v16h bz = load_b_frag(Wz, DHID, kk, lane);
        v16h bn = load_b_frag(Wn, DHID, kk, lane);
#pragma unroll
        for (int rt = 0; rt < 4; ++rt) {
          v16h a = load_a_frag(&hh[cur][rt * 16 * DHID], DHID, kk, lane);
          accR[rt] = WMMA_F16(a, br, accR[rt]);
          accZ[rt] = WMMA_F16(a, bz, accZ[rt]);
          accN[rt] = WMMA_F16(a, bn, accN[rt]);
        }
      }

      // GRU elementwise update directly on the WMMA C layout.
      const float bhr = b_hh[d0 + c];
      const float bhz = b_hh[DHID + d0 + c];
      const float bhn = b_hh[2 * DHID + d0 + c];
#pragma unroll
      for (int rt = 0; rt < 4; ++rt) {
#pragma unroll
        for (int r = 0; r < 8; ++r) {
          const int m    = rt * 16 + rbase + r;  // block-local row
          const int lsrc = l0 + m + shift;       // window source position
          float gxr = 0.f, gxz = 0.f, gxn = 0.f;
          if (lsrc >= 0) {
            const size_t grow = (size_t)(bbase + lsrc) * GD + d0 + c;
            gxr = gx[grow];
            gxz = gx[grow + DHID];
            gxn = gx[grow + 2 * DHID];
          }
          const float rg = fsigmoid(gxr + accR[rt][r] + bhr);
          const float zg = fsigmoid(gxz + accZ[rt][r] + bhz);
          const float ng = tanhf(gxn + rg * (accN[rt][r] + bhn));
          const int  hidx = m * DHID + d0 + c;
          const float hp = hf[hidx];
          const float hn = (1.f - zg) * ng + zg * hp;
          hf[hidx]      = hn;
          hh[nxt][hidx] = (_Float16)hn;
        }
      }
    }
    __syncthreads();   // all hh[nxt] writes done; hh[cur] reads done
    cur = nxt;
  }

  for (int i = tid; i < MTILE * DHID; i += 256)
    out[(size_t)m0 * DHID + i] = hf[i];
}

// ---------------------------------------------------------------------------
extern "C" void kernel_launch(void* const* d_in, const int* in_sizes, int n_in,
                              void* d_out, int out_size, void* d_ws, size_t ws_size,
                              hipStream_t stream) {
  const float* x    = (const float*)d_in[0];  // [8,1024,512]
  const float* W_ih = (const float*)d_in[1];  // [1536,512]
  const float* W_hh = (const float*)d_in[2];  // [1536,512]
  const float* b_ih = (const float*)d_in[3];  // [1536]
  const float* b_hh = (const float*)d_in[4];  // [1536]
  float* out = (float*)d_out;                 // [8,1024,512]

  _Float16* xh  = (_Float16*)d_ws;                      // 8192*512 f16  (8 MB)
  _Float16* wih = xh  + (size_t)NROWS * DHID;           // 1536*512 f16
  _Float16* whh = wih + (size_t)GD * DHID;              // 1536*512 f16
  float*    gx  = (float*)(whh + (size_t)GD * DHID);    // 8192*1536 f32 (48 MB)

  const int nx = NROWS * DHID;
  const int nw = GD * DHID;
  cvt_f32_f16<<<(nx + 255) / 256, 256, 0, stream>>>(x, xh, nx);
  cvt_f32_f16<<<(nw + 255) / 256, 256, 0, stream>>>(W_ih, wih, nw);
  cvt_f32_f16<<<(nw + 255) / 256, 256, 0, stream>>>(W_hh, whh, nw);

  dim3 gxgrid(GD / 64, NROWS / 16);
  gx_gemm<<<gxgrid, 128, 0, stream>>>(xh, wih, b_ih, gx);

  local_gru<<<NROWS / MTILE, 256, 0, stream>>>(whh, b_hh, gx, out);
}
